// GCNClassifier_Linear_80642305949904
// MI455X (gfx1250) — compile-verified, matrix operand order checked by reading, not run on hardware
//
#include <hip/hip_runtime.h>
#include <hip/hip_bf16.h>

// ---------------------------------------------------------------------------
// GCN classifier forward for MI455X (gfx1250, wave32, WMMA).
//
// Pipeline (per kernel_launch call, all on `stream`):
//   1) CSR build       : in-degree histogram (int atomics), 2-level exclusive
//                        scan -> rowptr, fill -> esrc[] (edges grouped by dst)
//   2) dinv            : dinv[i] = rsqrt(1 + indeg(i))
//   3) convert         : X -> bf16 (pad rows zeroed), W1,W2 -> bf16 transposed
//   4) GEMM1 (WMMA)    : Hs1 = (X@W1) * dinv[row]        (bf16 in, f32 acc)
//   5) gather1+final   : H1bf[i] = bf16(relu(dinv[i]*(Hs1[i] + sum_e Hs1[src]) + b1))
//   6) GEMM2 (WMMA)    : Hs2 = (H1@W2) * dinv[row]
//   7) gather2+final   : h2[i] = dinv[i]*(Hs2[i] + sum_e Hs2[src]) + b2
//   8) pool            : pool[g] += h2[i]   (sorted batch, register accum)
//   9) head            : out = relu(relu(relu(pool/cnt) @ Wf1 + bf1) @ Wf2 + bf2)
//
// No float atomics on the hot path: message passing is a CSR gather with
// coalesced reads + plain stores (atomics would cost ~384M fadd ops).
//
// WMMA layout (CDNA5 ISA 7.12.2, 16-bit A 16x32):
//   lane l: m = l&15, h = l>>4; fragment = A[m][k0+8h..+8) ++ A[m][k0+16+8h..+8)
//   -> two contiguous global_load_b128 per fragment; B symmetric with
//   N in place of M (B kept column-major: Wt[n][k], K-contiguous).
// ---------------------------------------------------------------------------

typedef __bf16 bf16_t;
typedef bf16_t v16bf __attribute__((ext_vector_type(16)));
typedef bf16_t v8bf  __attribute__((ext_vector_type(8)));
typedef float  v8f   __attribute__((ext_vector_type(8)));

constexpr int DIN  = 256;
constexpr int DH   = 512;
constexpr int DH2  = 256;   // DH/2
constexpr int DH4  = 128;   // DH/4
constexpr int DOUT = 16;
constexpr int NG   = 64;

__device__ __forceinline__ void atomAddF(float* p, float v) {
  unsafeAtomicAdd(p, v);          // global_atomic_add_f32
}

__device__ __forceinline__ v16bf load_frag(const bf16_t* __restrict__ row, int k0, int h) {
  union { v16bf v; v8bf half[2]; } u;
  u.half[0] = *(const v8bf*)(row + k0 + h * 8);        // K = k0+8h   .. +8
  u.half[1] = *(const v8bf*)(row + k0 + 16 + h * 8);   // K = k0+16+8h.. +8
  return u.v;
}

#define WMMA_BF16(A, Bv, C) \
  __builtin_amdgcn_wmma_f32_16x16x32_bf16(false, (A), false, (Bv), (short)0, (C), false, false)

// ------------------------------------------------------------- CSR build ----
__global__ void k_zero_int(int* p, int n) {
  int i = blockIdx.x * blockDim.x + threadIdx.x;
  if (i < n) p[i] = 0;
}
__global__ void k_count(const int* __restrict__ dst, int* cnt, int e) {
  int i = blockIdx.x * blockDim.x + threadIdx.x;
  if (i < e) atomicAdd(&cnt[dst[i]], 1);
}
// Hillis-Steele exclusive scan, 256-wide blocks, two levels (N <= 65536).
__global__ __launch_bounds__(256) void k_scan_block(const int* __restrict__ cnt,
                                                    int* __restrict__ rowp,
                                                    int* __restrict__ bsum, int n) {
  __shared__ int sh[256];
  int i = blockIdx.x * 256 + threadIdx.x;
  int v = (i < n) ? cnt[i] : 0;
  sh[threadIdx.x] = v;
  __syncthreads();
  for (int d = 1; d < 256; d <<= 1) {
    int t = (threadIdx.x >= d) ? sh[threadIdx.x - d] : 0;
    __syncthreads();
    sh[threadIdx.x] += t;
    __syncthreads();
  }
  if (i < n) rowp[i] = sh[threadIdx.x] - v;          // exclusive
  if (threadIdx.x == 255) bsum[blockIdx.x] = sh[255];
}
__global__ __launch_bounds__(256) void k_scan_sums(int* bsum, int nb) {
  __shared__ int sh[256];
  int v = (threadIdx.x < nb) ? bsum[threadIdx.x] : 0;
  sh[threadIdx.x] = v;
  __syncthreads();
  for (int d = 1; d < 256; d <<= 1) {
    int t = (threadIdx.x >= d) ? sh[threadIdx.x - d] : 0;
    __syncthreads();
    sh[threadIdx.x] += t;
    __syncthreads();
  }
  if (threadIdx.x < nb) bsum[threadIdx.x] = sh[threadIdx.x] - v;   // exclusive
}
__global__ void k_scan_add(int* rowp, const int* __restrict__ bsum, int n, int e) {
  int i = blockIdx.x * 256 + threadIdx.x;
  if (i < n) rowp[i] += bsum[blockIdx.x];
  if (i == 0) rowp[n] = e;
}
__global__ void k_fill(const int* __restrict__ src, const int* __restrict__ dst,
                       const int* __restrict__ rowp, int* fill,
                       int* __restrict__ esrc, int e) {
  int i = blockIdx.x * blockDim.x + threadIdx.x;
  if (i < e) {
    int d = dst[i];
    int pos = atomicAdd(&fill[d], 1);
    esrc[rowp[d] + pos] = src[i];
  }
}
__global__ void k_dinv(const int* __restrict__ cnt, float* dinv, int n) {
  int i = blockIdx.x * blockDim.x + threadIdx.x;
  if (i < n) dinv[i] = rsqrtf(1.0f + (float)cnt[i]);   // +1 self-loop
}

// -------------------------------------------------------------- converts ----
__global__ void k_cvt_x(const float* __restrict__ x, bf16_t* __restrict__ xb,
                        int nrows, int mpad) {
  size_t t = (size_t)blockIdx.x * blockDim.x + threadIdx.x;
  if (t >= (size_t)mpad * DIN) return;
  int i = (int)(t >> 8);                                // DIN = 256
  xb[t] = (i < nrows) ? (bf16_t)x[t] : (bf16_t)0.0f;
}
// W:[K,N] row-major f32  ->  Wt:[N,K] bf16 (K-contiguous per output column)
__global__ void k_cvt_wt(const float* __restrict__ W, bf16_t* __restrict__ Wt,
                         int K, int N) {
  size_t t = (size_t)blockIdx.x * blockDim.x + threadIdx.x;
  if (t >= (size_t)K * N) return;
  int n = (int)(t / K), k = (int)(t % K);
  Wt[t] = (bf16_t)W[(size_t)k * N + n];
}

// ------------------------------------------------------------ WMMA GEMM -----
// Per wave: 16 rows x 64 cols; block = 8 waves -> 128 x 64 tile.
// Register double-buffered K-loop: fragments for step k+32 are issued before
// the WMMAs of step k so each v_wmma waits on loads launched a step earlier.
template <int K, int NOUT>
__global__ __launch_bounds__(256) void k_gemm(
    const bf16_t* __restrict__ A, const bf16_t* __restrict__ Bt,
    float* __restrict__ Hs, const float* __restrict__ dinv, int nrows) {
  const int lane = threadIdx.x & 31;
  const int wid  = threadIdx.x >> 5;
  const int l16  = lane & 15;
  const int h    = lane >> 4;
  const int row0 = blockIdx.x * 128 + wid * 16;
  const int col0 = blockIdx.y * 64;

  const bf16_t* arow = A  + (size_t)(row0 + l16) * K;
  const bf16_t* b0r  = Bt + (size_t)(col0 +  0 + l16) * K;
  const bf16_t* b1r  = Bt + (size_t)(col0 + 16 + l16) * K;
  const bf16_t* b2r  = Bt + (size_t)(col0 + 32 + l16) * K;
  const bf16_t* b3r  = Bt + (size_t)(col0 + 48 + l16) * K;

  v16bf a  = load_frag(arow, 0, h);
  v16bf b0 = load_frag(b0r,  0, h);
  v16bf b1 = load_frag(b1r,  0, h);
  v16bf b2 = load_frag(b2r,  0, h);
  v16bf b3 = load_frag(b3r,  0, h);

  v8f acc0 = {}, acc1 = {}, acc2 = {}, acc3 = {};
#pragma unroll
  for (int k0 = 0; k0 < K; k0 += 32) {
    if (k0 + 32 < K) {
      v16bf an  = load_frag(arow, k0 + 32, h);
      v16bf bn0 = load_frag(b0r,  k0 + 32, h);
      v16bf bn1 = load_frag(b1r,  k0 + 32, h);
      v16bf bn2 = load_frag(b2r,  k0 + 32, h);
      v16bf bn3 = load_frag(b3r,  k0 + 32, h);
      acc0 = WMMA_BF16(a, b0, acc0);
      acc1 = WMMA_BF16(a, b1, acc1);
      acc2 = WMMA_BF16(a, b2, acc2);
      acc3 = WMMA_BF16(a, b3, acc3);
      a = an; b0 = bn0; b1 = bn1; b2 = bn2; b3 = bn3;
    } else {
      acc0 = WMMA_BF16(a, b0, acc0);
      acc1 = WMMA_BF16(a, b1, acc1);
      acc2 = WMMA_BF16(a, b2, acc2);
      acc3 = WMMA_BF16(a, b3, acc3);
    }
  }

  // C/D layout: lane l, vgpr v -> row = 8*(l>>4)+v, col = l&15 (per 16-col tile)
  const int rbase = row0 + h * 8;
#pragma unroll
  for (int v = 0; v < 8; ++v) {
    int r = rbase + v;
    if (r < nrows) {
      float dv = dinv[r];
      size_t o = (size_t)r * NOUT + col0 + l16;
      Hs[o +  0] = acc0[v] * dv;
      Hs[o + 16] = acc1[v] * dv;
      Hs[o + 32] = acc2[v] * dv;
      Hs[o + 48] = acc3[v] * dv;
    }
  }
}

// ------------------------------------------------- CSR gather + finalize ----
// One block per destination row; coalesced feature reads, plain stores.
// conv1: F=512, thread covers f and f+256; fused bias+relu+bf16 convert.
__global__ __launch_bounds__(256) void k_gather1(
    const float* __restrict__ Hs, const int* __restrict__ rowp,
    const int* __restrict__ esrc, const float* __restrict__ dinv,
    const float* __restrict__ b1, bf16_t* __restrict__ H1) {
  int i = blockIdx.x;
  int f = threadIdx.x;
  const float* self = Hs + (size_t)i * DH;
  float a0 = self[f], a1 = self[f + 256];       // self-loop term (pre-scaled)
  int e0 = rowp[i], e1 = rowp[i + 1];
  for (int e = e0; e < e1; ++e) {
    const float* srow = Hs + (size_t)esrc[e] * DH;
    a0 += srow[f];
    a1 += srow[f + 256];
  }
  float dv = dinv[i];
  size_t o = (size_t)i * DH;
  H1[o + f]       = (bf16_t)fmaxf(dv * a0 + b1[f], 0.0f);
  H1[o + f + 256] = (bf16_t)fmaxf(dv * a1 + b1[f + 256], 0.0f);
}
// conv2: F=256, one feature per thread; fused bias (no relu before pooling).
__global__ __launch_bounds__(256) void k_gather2(
    const float* __restrict__ Hs, const int* __restrict__ rowp,
    const int* __restrict__ esrc, const float* __restrict__ dinv,
    const float* __restrict__ b2, float* __restrict__ h2) {
  int i = blockIdx.x;
  int f = threadIdx.x;
  float a = Hs[(size_t)i * DH2 + f];
  int e0 = rowp[i], e1 = rowp[i + 1];
  for (int e = e0; e < e1; ++e) a += Hs[(size_t)esrc[e] * DH2 + f];
  h2[(size_t)i * DH2 + f] = dinv[i] * a + b2[f];
}
__global__ void k_zero_padH1(bf16_t* H1, int nrows, int mpad) {
  size_t t = (size_t)blockIdx.x * blockDim.x + threadIdx.x;
  size_t n = (size_t)(mpad - nrows) * DH;
  if (t < n) H1[(size_t)nrows * DH + t] = (bf16_t)0.0f;
}

// ----------------------------------------------------------------- pool -----
__global__ void k_zero_pool(float* pool, float* cnt) {
  int t = blockIdx.x * blockDim.x + threadIdx.x;
  if (t < NG * DH2) pool[t] = 0.0f;
  if (t < NG) cnt[t] = 0.0f;
}
__global__ void k_cnt(const int* __restrict__ batch, float* cnt, int n) {
  int i = blockIdx.x * blockDim.x + threadIdx.x;
  if (i < n) atomAddF(&cnt[batch[i]], 1.0f);
}
// 256 threads (one per feature), 64 sorted nodes per block; register
// accumulation with atomic flush only at graph boundaries.
__global__ __launch_bounds__(256) void k_pool(
    const float* __restrict__ h2, const int* __restrict__ batch,
    float* __restrict__ pool, int nrows) {
  int f    = threadIdx.x;
  int base = blockIdx.x * 64;
  float acc = 0.0f;
  int cur = -1;
  for (int r = 0; r < 64; ++r) {
    int i = base + r;
    if (i >= nrows) break;
    int g = batch[i];
    if (g != cur) {
      if (cur >= 0) atomAddF(&pool[cur * DH2 + f], acc);
      cur = g; acc = 0.0f;
    }
    acc += h2[(size_t)i * DH2 + f];
  }
  if (cur >= 0) atomAddF(&pool[cur * DH2 + f], acc);
}

// ----------------------------------------------------------------- head -----
__global__ __launch_bounds__(128) void k_head1(
    const float* __restrict__ pool, const float* __restrict__ cnt,
    const float* __restrict__ Wf1, const float* __restrict__ bf1,
    float* __restrict__ g1) {
  __shared__ float gsh[DH2];
  int gi = blockIdx.x;
  float ic = 1.0f / fmaxf(cnt[gi], 1.0f);
  for (int f = threadIdx.x; f < DH2; f += 128)
    gsh[f] = fmaxf(pool[gi * DH2 + f] * ic, 0.0f);     // mean then relu
  __syncthreads();
  int j = threadIdx.x;
  float s = bf1[j];
  for (int k = 0; k < DH2; ++k) s += gsh[k] * Wf1[k * DH4 + j];
  g1[gi * DH4 + j] = fmaxf(s, 0.0f);
}
__global__ __launch_bounds__(128) void k_head2(
    const float* __restrict__ g1, const float* __restrict__ Wf2,
    const float* __restrict__ bf2, float* __restrict__ out) {
  __shared__ float g1sh[DH4];
  int gi = blockIdx.x;
  g1sh[threadIdx.x] = g1[gi * DH4 + threadIdx.x];
  __syncthreads();
  if (threadIdx.x < DOUT) {
    int j = threadIdx.x;
    float s = bf2[j];
    for (int k = 0; k < DH4; ++k) s += g1sh[k] * Wf2[k * DOUT + j];
    out[gi * DOUT + j] = fmaxf(s, 0.0f);
  }
}

// ---------------------------------------------------------------------------
static inline size_t cdiv(size_t a, size_t b) { return (a + b - 1) / b; }

extern "C" void kernel_launch(void* const* d_in, const int* in_sizes, int n_in,
                              void* d_out, int out_size, void* d_ws, size_t ws_size,
                              hipStream_t stream) {
  const float* x   = (const float*)d_in[0];
  const float* W1  = (const float*)d_in[1];
  const float* b1  = (const float*)d_in[2];
  const float* W2  = (const float*)d_in[3];
  const float* b2  = (const float*)d_in[4];
  const float* Wf1 = (const float*)d_in[5];
  const float* bf1 = (const float*)d_in[6];
  const float* Wf2 = (const float*)d_in[7];
  const float* bf2 = (const float*)d_in[8];
  const int* eidx  = (const int*)d_in[9];
  const int* batch = (const int*)d_in[10];

  const int N = in_sizes[10];            // nodes
  const int E = in_sizes[9] / 2;         // edges
  const int Mpad = (int)cdiv(N, 128) * 128;
  const int* src = eidx;
  const int* dst = eidx + E;

  // workspace carve-out (256B aligned regions)
  char* base = (char*)d_ws;
  size_t off = 0;
  auto carve = [&](size_t bytes) -> char* {
    char* p = base + off;
    off = (off + bytes + 255) & ~(size_t)255;
    return p;
  };
  float*  dinv  = (float*)carve((size_t)N * 4);
  int*    cnt32 = (int*)carve((size_t)N * 4);           // histogram, then fill cursor
  int*    rowp  = (int*)carve((size_t)(N + 1) * 4);
  int*    bsum  = (int*)carve(256 * 4);
  int*    esrc  = (int*)carve((size_t)E * 4);
  bf16_t* Xbf   = (bf16_t*)carve((size_t)Mpad * DIN * 2);
  bf16_t* W1t   = (bf16_t*)carve((size_t)DH * DIN * 2);
  bf16_t* W2t   = (bf16_t*)carve((size_t)DH2 * DH * 2);
  bf16_t* H1bf  = (bf16_t*)carve((size_t)Mpad * DH * 2);
  float*  HsR   = (float*)carve((size_t)Mpad * DH * 4); // Hs1 / Hs2 (aliased)
  float*  h2    = (float*)carve((size_t)N * DH2 * 4);
  float*  pool  = (float*)carve((size_t)NG * DH2 * 4);
  float*  cntf  = (float*)carve((size_t)NG * 4);
  float*  g1b   = (float*)carve((size_t)NG * DH4 * 4);
  (void)ws_size; (void)n_in; (void)out_size;

  const int B  = 256;
  const int nb = (int)cdiv(N, 256);      // <= 256 for N <= 65536

  // 1) CSR build + dinv
  k_zero_int<<<cdiv(N, B), B, 0, stream>>>(cnt32, N);
  k_count<<<cdiv(E, B), B, 0, stream>>>(dst, cnt32, E);
  k_scan_block<<<nb, 256, 0, stream>>>(cnt32, rowp, bsum, N);
  k_scan_sums<<<1, 256, 0, stream>>>(bsum, nb);
  k_scan_add<<<nb, 256, 0, stream>>>(rowp, bsum, N, E);
  k_dinv<<<cdiv(N, B), B, 0, stream>>>(cnt32, dinv, N);
  k_zero_int<<<cdiv(N, B), B, 0, stream>>>(cnt32, N);   // reuse as fill cursor
  k_fill<<<cdiv(E, B), B, 0, stream>>>(src, dst, rowp, cnt32, esrc, E);

  // 2) converts
  k_cvt_x<<<cdiv((size_t)Mpad * DIN, B), B, 0, stream>>>(x, Xbf, N, Mpad);
  k_cvt_wt<<<cdiv((size_t)DIN * DH, B), B, 0, stream>>>(W1, W1t, DIN, DH);
  k_cvt_wt<<<cdiv((size_t)DH * DH2, B), B, 0, stream>>>(W2, W2t, DH, DH2);

  // 3) conv1: GEMM + gather/finalize
  {
    dim3 g(Mpad / 128, DH / 64);
    k_gemm<DIN, DH><<<g, B, 0, stream>>>(Xbf, W1t, HsR, dinv, N);
  }
  k_gather1<<<N, 256, 0, stream>>>(HsR, rowp, esrc, dinv, b1, H1bf);
  k_zero_padH1<<<cdiv((size_t)(Mpad - N) * DH, B), B, 0, stream>>>(H1bf, N, Mpad);

  // 4) conv2: GEMM + gather/finalize
  {
    dim3 g(Mpad / 128, DH2 / 64);
    k_gemm<DH, DH2><<<g, B, 0, stream>>>(H1bf, W2t, HsR, dinv, N);
  }
  k_gather2<<<N, 256, 0, stream>>>(HsR, rowp, esrc, dinv, b2, h2);

  // 5) pool
  k_zero_pool<<<cdiv(NG * DH2, B), B, 0, stream>>>(pool, cntf);
  k_cnt<<<cdiv(N, B), B, 0, stream>>>(batch, cntf, N);
  k_pool<<<cdiv(N, 64), 256, 0, stream>>>(h2, batch, pool, N);

  // 6) head
  k_head1<<<NG, 128, 0, stream>>>(pool, cntf, Wf1, bf1, g1b);
  k_head2<<<NG, 128, 0, stream>>>(g1b, Wf2, bf2, (float*)d_out);
}
